// MultiHeadAttention_48679159333494
// MI455X (gfx1250) — matrix operand order repaired and checked by reading, not compiled, not run on hardware
//
#include <hip/hip_runtime.h>
#include <hip/hip_bf16.h>

// ---------------------------------------------------------------------------
// Multi-head attention for MI455X (gfx1250, wave32, WMMA + TDM).
//   B=2, S=2048, E=1024, H=16, D=64
// Pipeline:
//   k1 (x3): Q/K/V = x @ W^T   f32 -> bf16; Q,K as [B*H, S, D], V as [B*H, D, S]
//   k2     : flash attention   bf16 -> bf16 ctx [B, S, E]
//            K/V tiles staged by the Tensor Data Mover (tensor_load_to_lds,
//            TENSORcnt-tracked, LDS pad field reproduces the 72-ushort stride)
//   k3     : out = ctx @ Wo^T  bf16 -> f32 out [B, S, E]
// All GEMM-shaped math: v_wmma_f32_16x16x32_bf16 (f32 accumulate).
// ---------------------------------------------------------------------------

#define S_LEN 2048
#define EMB   1024
#define NH    16
#define DHEAD 64
#define NB    2
#define NROWS (NB * S_LEN) // 4096

typedef __attribute__((ext_vector_type(16))) __bf16        v16bf;
typedef __attribute__((ext_vector_type(8)))  float         v8f;
typedef __attribute__((ext_vector_type(4)))  unsigned int  u32x4;
typedef __attribute__((ext_vector_type(8)))  int           i32x8;
typedef __attribute__((ext_vector_type(4)))  int           i32x4;

#if __has_builtin(__builtin_amdgcn_tensor_load_to_lds) && \
    __has_builtin(__builtin_amdgcn_s_wait_tensorcnt)
#define USE_TDM 1
#else
#define USE_TDM 0
#endif

union FragBF { u32x4 u[2]; v16bf v; };

__device__ __forceinline__ unsigned pack2bf(float lo, float hi) {
  unsigned ul = __builtin_bit_cast(unsigned, lo) + 0x8000u;
  unsigned uh = __builtin_bit_cast(unsigned, hi) + 0x8000u;
#if __has_builtin(__builtin_amdgcn_perm)
  return __builtin_amdgcn_perm(uh, ul, 0x07060302u);
#else
  return (uh & 0xFFFF0000u) | (ul >> 16);
#endif
}

__device__ __forceinline__ unsigned short f2bf(float f) {
  unsigned u = __builtin_bit_cast(unsigned, f) + 0x8000u;
  return (unsigned short)(u >> 16);
}

__device__ __forceinline__ u32x4 packf4x2(float4 a, float4 b) {
  u32x4 r;
  r.x = pack2bf(a.x, a.y); r.y = pack2bf(a.z, a.w);
  r.z = pack2bf(b.x, b.y); r.w = pack2bf(b.z, b.w);
  return r;
}

__device__ __forceinline__ float half16_max(float v) {
  v = fmaxf(v, __shfl_xor(v, 1));
  v = fmaxf(v, __shfl_xor(v, 2));
  v = fmaxf(v, __shfl_xor(v, 4));
  v = fmaxf(v, __shfl_xor(v, 8));
  return v;
}
__device__ __forceinline__ float half16_sum(float v) {
  v += __shfl_xor(v, 1);
  v += __shfl_xor(v, 2);
  v += __shfl_xor(v, 4);
  v += __shfl_xor(v, 8);
  return v;
}

#if USE_TDM
// TDM load of a 64x64 bf16 tile into LDS with +16B row padding (stride 144B).
// D# per CDNA5 ISA ch.8: group0 = {count/lds_addr/global_addr/type},
// group1 = {flags, dims, tile dims, dim0 stride}. 2D tile -> groups 2/3 zero.
__device__ __forceinline__ void tdm_load_tile64x64(
    unsigned ldsOff, const unsigned short* gsrc, unsigned strideElems) {
  const unsigned long long ga = (unsigned long long)(size_t)gsrc;
  u32x4 g0;
  g0.x = 1u;                                             // count=1 (valid)
  g0.y = ldsOff;                                         // lds_addr bytes
  g0.z = (unsigned)(ga & 0xFFFFFFFFull);                 // global_addr[31:0]
  g0.w = (unsigned)((ga >> 32) & 0x01FFFFFFull) | (2u << 30); // addr[56:32]|type=2
  i32x8 g1;
  g1[0] = (1 << 16)        // data_size = 2 bytes
        | (1 << 20)        // pad_enable
        | (4 << 22)        // pad_interval: 32 DWORDs (=128B, one tile row)
        | (3 << 25);       // pad_amount: 4 DWORDs (=16B) -> 144B LDS row
  g1[1] = (64 << 16);      // tensor_dim0 = 64
  g1[2] = (64 << 16);      // tensor_dim1 = 64
  g1[3] = (64 << 16);      // tile_dim0 = 64
  g1[4] = 64;              // tile_dim1 = 64, tile_dim2 = 0
  g1[5] = (int)strideElems;// tensor_dim0_stride (elements)
  g1[6] = 0;
  g1[7] = 0;
  i32x4 z4 = {};
#if __clang_major__ >= 23
  i32x8 z8 = {};
  __builtin_amdgcn_tensor_load_to_lds(g0, g1, z4, z4, z8, 0);
#else
  __builtin_amdgcn_tensor_load_to_lds(g0, g1, z4, z4, 0);
#endif
}
#endif

// ---------------------------------------------------------------------------
// Kernel 1: out = x @ W^T, bf16. vT=0: [b*H+h][s][d]; vT=1: [b*H+h][d][s]
// 256 threads (8 waves), 128x128 tile, K step 32, double-buffered LDS.
// ---------------------------------------------------------------------------
__global__ __launch_bounds__(256) void qkv_proj_kernel(
    const float* __restrict__ x, const float* __restrict__ W,
    unsigned short* __restrict__ out, const int vT) {
  __shared__ __align__(16) unsigned short lA[2][128 * 40];
  __shared__ __align__(16) unsigned short lB[2][128 * 40];

  const int tid  = threadIdx.x;
  const int wave = tid >> 5;
  const int lane = tid & 31;
  const int hh   = lane >> 4;
  const int lr   = lane & 15;

  const int rbase = blockIdx.x * 128;
  const int cbase = blockIdx.y * 128;

  const int srow = tid >> 1;
  const int skh  = (tid & 1) * 16;

  const float* aSrc = x + (size_t)(rbase + srow) * EMB + skh;
  const float* bSrc = W + (size_t)(cbase + srow) * EMB + skh;

  v8f acc[8] = {};
  float4 ra[4], rb[4];

  {
    const float4* a4 = (const float4*)aSrc;
    const float4* b4 = (const float4*)bSrc;
#pragma unroll
    for (int j = 0; j < 4; ++j) { ra[j] = a4[j]; rb[j] = b4[j]; }
    *(u32x4*)&lA[0][srow*40 + skh]     = packf4x2(ra[0], ra[1]);
    *(u32x4*)&lA[0][srow*40 + skh + 8] = packf4x2(ra[2], ra[3]);
    *(u32x4*)&lB[0][srow*40 + skh]     = packf4x2(rb[0], rb[1]);
    *(u32x4*)&lB[0][srow*40 + skh + 8] = packf4x2(rb[2], rb[3]);
  }

  int p = 0;
  for (int k0 = 0; k0 < EMB; k0 += 32) {
    const bool nxt = (k0 + 32) < EMB;
    if (nxt) {
      const float4* a4 = (const float4*)(aSrc + k0 + 32);
      const float4* b4 = (const float4*)(bSrc + k0 + 32);
#pragma unroll
      for (int j = 0; j < 4; ++j) { ra[j] = a4[j]; rb[j] = b4[j]; }
      if (k0 + 64 < EMB) {
        __builtin_prefetch(aSrc + k0 + 64, 0, 0);
        __builtin_prefetch(bSrc + k0 + 64, 0, 0);
      }
    }
    __syncthreads();

    FragBF af;
    const int am = (wave * 16 + lr) * 40;
    af.u[0] = *(const u32x4*)&lA[p][am + 8*hh];
    af.u[1] = *(const u32x4*)&lA[p][am + 16 + 8*hh];
#pragma unroll
    for (int c = 0; c < 8; ++c) {
      FragBF bf;
      const int bn = (c * 16 + lr) * 40 + 8*hh;
      bf.u[0] = *(const u32x4*)&lB[p][bn];
      bf.u[1] = *(const u32x4*)&lB[p][bn + 16];
      acc[c] = __builtin_amdgcn_wmma_f32_16x16x32_bf16(
          false, af.v, false, bf.v, (short)0, acc[c], false, false);
    }

    if (nxt) {
      const int q = p ^ 1;
      *(u32x4*)&lA[q][srow*40 + skh]     = packf4x2(ra[0], ra[1]);
      *(u32x4*)&lA[q][srow*40 + skh + 8] = packf4x2(ra[2], ra[3]);
      *(u32x4*)&lB[q][srow*40 + skh]     = packf4x2(rb[0], rb[1]);
      *(u32x4*)&lB[q][srow*40 + skh + 8] = packf4x2(rb[2], rb[3]);
    }
    p ^= 1;
  }

#pragma unroll
  for (int c = 0; c < 8; ++c) {
    const int col = cbase + c * 16 + lr;
    const int h = col >> 6, d = col & 63;
#pragma unroll
    for (int r = 0; r < 8; ++r) {
      const int row = rbase + wave * 16 + hh * 8 + r;
      const int b = row >> 11, s = row & (S_LEN - 1);
      const unsigned short val = f2bf(acc[c][r]);
      if (vT) out[((size_t)(b * NH + h) * DHEAD + d) * S_LEN + s] = val;
      else    out[((size_t)(b * NH + h) * S_LEN + s) * DHEAD + d] = val;
    }
  }
}

// ---------------------------------------------------------------------------
// Kernel 2: flash attention, double-buffered K/V (TDM when available),
// V pre-transposed [b*H+h][d][s].
// ---------------------------------------------------------------------------
__global__ __launch_bounds__(128) void attn_kernel(
    const unsigned short* __restrict__ Q,
    const unsigned short* __restrict__ K,
    const unsigned short* __restrict__ Vt,
    unsigned short* __restrict__ ctx) {
  __shared__ __align__(16) unsigned short lK [2][64 * 72];
  __shared__ __align__(16) unsigned short lVt[2][64 * 72];
  __shared__ __align__(16) unsigned short lP [4 * 16 * 72];

  const int tid  = threadIdx.x;
  const int wave = tid >> 5;
  const int lane = tid & 31;
  const int hh   = lane >> 4;
  const int lr   = lane & 15;

  const int qtile = blockIdx.x;
  const int bh    = blockIdx.y;
  const size_t bhBase = (size_t)bh * S_LEN * DHEAD;
  const int qg0 = qtile * 64 + wave * 16;

  const unsigned short* qp = Q + bhBase + (size_t)(qg0 + lr) * DHEAD;
  FragBF qf[2];
#pragma unroll
  for (int t = 0; t < 2; ++t) {
    qf[t].u[0] = *(const u32x4*)(qp + 32*t + 8*hh);
    qf[t].u[1] = *(const u32x4*)(qp + 32*t + 16 + 8*hh);
  }

  float m_run[8], l_run[8];
#pragma unroll
  for (int r = 0; r < 8; ++r) { m_run[r] = -1e30f; l_run[r] = 0.f; }
  v8f oacc[4] = {};

  const unsigned short* kBase = K  + bhBase;  // [kv][d], row stride DHEAD
  const unsigned short* vBase = Vt + bhBase;  // [d][s],  row stride S_LEN

#if USE_TDM
  const unsigned ldsK0 = (unsigned)(size_t)&lK [0][0];
  const unsigned ldsK1 = (unsigned)(size_t)&lK [1][0];
  const unsigned ldsV0 = (unsigned)(size_t)&lVt[0][0];
  const unsigned ldsV1 = (unsigned)(size_t)&lVt[1][0];
  // prologue: wave0 DMAs K tile 0, wave1 DMAs V tile 0 (TDM is wave-level)
  if (wave == 0)      tdm_load_tile64x64(ldsK0, kBase, DHEAD);
  else if (wave == 1) tdm_load_tile64x64(ldsV0, vBase, S_LEN);
  if (wave < 2) __builtin_amdgcn_s_wait_tensorcnt(0);
#else
  const int srow = tid >> 1;
  const int sh32 = (tid & 1) * 32;
  const unsigned short* kp = kBase + (size_t)srow * DHEAD + sh32;
  const unsigned short* vp = vBase + (size_t)srow * S_LEN + sh32;
  u32x4 rk[4], rv[4];
  {
#pragma unroll
    for (int j = 0; j < 4; ++j) {
      rk[j] = *(const u32x4*)(kp + 8*j);
      rv[j] = *(const u32x4*)(vp + 8*j);
    }
#pragma unroll
    for (int j = 0; j < 4; ++j) {
      *(u32x4*)&lK [0][srow*72 + sh32 + 8*j] = rk[j];
      *(u32x4*)&lVt[0][srow*72 + sh32 + 8*j] = rv[j];
    }
  }
#endif

  int p = 0;
  for (int it = 0; it < S_LEN / 64; ++it) {
    const bool nxt = (it + 1) < S_LEN / 64;
#if !USE_TDM
    if (nxt) {
      const int kv0n = (it + 1) * 64;
#pragma unroll
      for (int j = 0; j < 4; ++j) {
        rk[j] = *(const u32x4*)(kp + (size_t)kv0n * DHEAD + 8*j);
        rv[j] = *(const u32x4*)(vp + kv0n + 8*j);
      }
    }
#endif
    __syncthreads();   // buffer p staged for all waves; buffer q free

#if USE_TDM
    if (nxt) {         // DMA next KV tile into buffer q; WMMAs below hide it
      const int kv0n = (it + 1) * 64;
      if (wave == 0)
        tdm_load_tile64x64((it & 1) ? ldsK0 : ldsK1,
                           kBase + (size_t)kv0n * DHEAD, DHEAD);
      else if (wave == 1)
        tdm_load_tile64x64((it & 1) ? ldsV0 : ldsV1,
                           vBase + kv0n, S_LEN);
    }
#endif

    // scores = Q @ K^T : B[k=d][n=kv] = Ktile[kv][d]
    v8f sacc[4] = {};
#pragma unroll
    for (int t = 0; t < 2; ++t) {
#pragma unroll
      for (int c = 0; c < 4; ++c) {
        FragBF kf;
        const int kr = (c * 16 + lr) * 72 + 32*t + 8*hh;
        kf.u[0] = *(const u32x4*)&lK[p][kr];
        kf.u[1] = *(const u32x4*)&lK[p][kr + 16];
        sacc[c] = __builtin_amdgcn_wmma_f32_16x16x32_bf16(
            false, qf[t].v, false, kf.v, (short)0, sacc[c], false, false);
      }
    }

    // online softmax; row m<8 in lanes 0-15, m>=8 in lanes 16-31
#pragma unroll
    for (int r = 0; r < 8; ++r) {
      float mx = -1e30f;
#pragma unroll
      for (int c = 0; c < 4; ++c) {
        sacc[c][r] *= 0.125f;                 // 1/sqrt(D)
        mx = fmaxf(mx, sacc[c][r]);
      }
      mx = half16_max(mx);
      const float newm = fmaxf(m_run[r], mx);
      const float corr = __expf(m_run[r] - newm);
      float ssum = 0.f;
#pragma unroll
      for (int c = 0; c < 4; ++c) {
        const float pv = __expf(sacc[c][r] - newm);
        sacc[c][r] = pv;
        ssum += pv;
      }
      ssum = half16_sum(ssum);
      m_run[r] = newm;
      l_run[r] = l_run[r] * corr + ssum;
#pragma unroll
      for (int c = 0; c < 4; ++c) oacc[c][r] *= corr;
    }

    // P (C-layout f32) -> bf16 A-layout via per-wave LDS scratch
    const int pb = wave * 16 * 72;
#pragma unroll
    for (int c = 0; c < 4; ++c)
#pragma unroll
      for (int r = 0; r < 8; ++r)
        lP[pb + (hh * 8 + r) * 72 + c * 16 + lr] = f2bf(sacc[c][r]);

    // O += P @ V : B[k=kv][n=d] = lVt[d][kv]
#pragma unroll
    for (int t = 0; t < 2; ++t) {
      FragBF pf;
      const int pr = pb + lr * 72 + 32*t;
      pf.u[0] = *(const u32x4*)&lP[pr + 8*hh];
      pf.u[1] = *(const u32x4*)&lP[pr + 16 + 8*hh];
#pragma unroll
      for (int c = 0; c < 4; ++c) {
        FragBF vf;
        const int vr = (c * 16 + lr) * 72 + 32*t + 8*hh;
        vf.u[0] = *(const u32x4*)&lVt[p][vr];
        vf.u[1] = *(const u32x4*)&lVt[p][vr + 16];
        oacc[c] = __builtin_amdgcn_wmma_f32_16x16x32_bf16(
            false, pf.v, false, vf.v, (short)0, oacc[c], false, false);
      }
    }

#if USE_TDM
    if (nxt && wave < 2) __builtin_amdgcn_s_wait_tensorcnt(0);
#else
    if (nxt) {
      const int q = p ^ 1;
#pragma unroll
      for (int j = 0; j < 4; ++j) {
        *(u32x4*)&lK [q][srow*72 + sh32 + 8*j] = rk[j];
        *(u32x4*)&lVt[q][srow*72 + sh32 + 8*j] = rv[j];
      }
    }
#endif
    p ^= 1;
  }

  const int b = bh >> 4, h = bh & 15;
#pragma unroll
  for (int r = 0; r < 8; ++r) {
    const float inv = 1.f / l_run[r];
    const int s = qg0 + hh * 8 + r;
#pragma unroll
    for (int c = 0; c < 4; ++c) {
      const int d = c * 16 + lr;
      ctx[((size_t)b * S_LEN + s) * EMB + h * DHEAD + d] = f2bf(oacc[c][r] * inv);
    }
  }
}

// ---------------------------------------------------------------------------
// Kernel 3: out_f32 = ctx_bf16 @ Wo^T.
// ---------------------------------------------------------------------------
__global__ __launch_bounds__(256) void out_proj_kernel(
    const unsigned short* __restrict__ ctx, const float* __restrict__ W,
    float* __restrict__ out) {
  __shared__ __align__(16) unsigned short lA[2][128 * 40];
  __shared__ __align__(16) unsigned short lB[2][128 * 40];

  const int tid  = threadIdx.x;
  const int wave = tid >> 5;
  const int lane = tid & 31;
  const int hh   = lane >> 4;
  const int lr   = lane & 15;

  const int rbase = blockIdx.x * 128;
  const int cbase = blockIdx.y * 128;

  const int srow = tid >> 1;
  const int skh  = (tid & 1) * 16;

  const unsigned short* aSrc = ctx + (size_t)(rbase + srow) * EMB + skh;
  const float*          bSrc = W   + (size_t)(cbase + srow) * EMB + skh;

  v8f acc[8] = {};
  u32x4 raw[2]; float4 rb[4];

  {
    raw[0] = *(const u32x4*)(aSrc);
    raw[1] = *(const u32x4*)(aSrc + 8);
    const float4* b4 = (const float4*)bSrc;
#pragma unroll
    for (int j = 0; j < 4; ++j) rb[j] = b4[j];
    *(u32x4*)&lA[0][srow*40 + skh]     = raw[0];
    *(u32x4*)&lA[0][srow*40 + skh + 8] = raw[1];
    *(u32x4*)&lB[0][srow*40 + skh]     = packf4x2(rb[0], rb[1]);
    *(u32x4*)&lB[0][srow*40 + skh + 8] = packf4x2(rb[2], rb[3]);
  }

  int p = 0;
  for (int k0 = 0; k0 < EMB; k0 += 32) {
    const bool nxt = (k0 + 32) < EMB;
    if (nxt) {
      raw[0] = *(const u32x4*)(aSrc + k0 + 32);
      raw[1] = *(const u32x4*)(aSrc + k0 + 40);
      const float4* b4 = (const float4*)(bSrc + k0 + 32);
#pragma unroll
      for (int j = 0; j < 4; ++j) rb[j] = b4[j];
      if (k0 + 64 < EMB) {
        __builtin_prefetch(aSrc + k0 + 64, 0, 0);
        __builtin_prefetch(bSrc + k0 + 64, 0, 0);
      }
    }
    __syncthreads();

    FragBF af;
    const int am = (wave * 16 + lr) * 40;
    af.u[0] = *(const u32x4*)&lA[p][am + 8*hh];
    af.u[1] = *(const u32x4*)&lA[p][am + 16 + 8*hh];
#pragma unroll
    for (int c = 0; c < 8; ++c) {
      FragBF bf;
      const int bn = (c * 16 + lr) * 40 + 8*hh;
      bf.u[0] = *(const u32x4*)&lB[p][bn];
      bf.u[1] = *(const u32x4*)&lB[p][bn + 16];
      acc[c] = __builtin_amdgcn_wmma_f32_16x16x32_bf16(
          false, af.v, false, bf.v, (short)0, acc[c], false, false);
    }

    if (nxt) {
      const int q = p ^ 1;
      *(u32x4*)&lA[q][srow*40 + skh]     = raw[0];
      *(u32x4*)&lA[q][srow*40 + skh + 8] = raw[1];
      *(u32x4*)&lB[q][srow*40 + skh]     = packf4x2(rb[0], rb[1]);
      *(u32x4*)&lB[q][srow*40 + skh + 8] = packf4x2(rb[2], rb[3]);
    }
    p ^= 1;
  }

#pragma unroll
  for (int c = 0; c < 8; ++c) {
    const int col = cbase + c * 16 + lr;
#pragma unroll
    for (int r = 0; r < 8; ++r) {
      const int row = rbase + wave * 16 + hh * 8 + r;
      out[(size_t)row * EMB + col] = acc[c][r];
    }
  }
}

// ---------------------------------------------------------------------------
extern "C" void kernel_launch(void* const* d_in, const int* in_sizes, int n_in,
                              void* d_out, int out_size, void* d_ws, size_t ws_size,
                              hipStream_t stream) {
  (void)in_sizes; (void)n_in; (void)out_size; (void)ws_size;
  const float* query = (const float*)d_in[0];
  const float* key   = (const float*)d_in[1];
  const float* value = (const float*)d_in[2];
  const float* Wq    = (const float*)d_in[3];
  const float* Wk    = (const float*)d_in[4];
  const float* Wv    = (const float*)d_in[5];
  const float* Wo    = (const float*)d_in[6];

  unsigned short* ws = (unsigned short*)d_ws;
  const size_t qkvN = (size_t)NB * NH * S_LEN * DHEAD;
  unsigned short* qws = ws;
  unsigned short* kws = qws + qkvN;
  unsigned short* vws = kws + qkvN;   // V stored transposed [b*H+h][d][s]
  unsigned short* cws = vws + qkvN;

  dim3 gProj(NROWS / 128, EMB / 128); // 32 x 8
  qkv_proj_kernel<<<gProj, 256, 0, stream>>>(query, Wq, qws, 0);
  qkv_proj_kernel<<<gProj, 256, 0, stream>>>(key,   Wk, kws, 0);
  qkv_proj_kernel<<<gProj, 256, 0, stream>>>(value, Wv, vws, 1);

  dim3 gAttn(S_LEN / 64, NB * NH);    // 32 x 32
  attn_kernel<<<gAttn, 128, 0, stream>>>(qws, kws, vws, cws);

  out_proj_kernel<<<gProj, 256, 0, stream>>>(cws, Wo, (float*)d_out);
}
